// Student_78675210928488
// MI455X (gfx1250) — compile-verified, matrix operand order checked by reading, not run on hardware
//
#include <hip/hip_runtime.h>
#include <hip/hip_bf16.h>

typedef __attribute__((ext_vector_type(16))) _Float16 v16h;
typedef __attribute__((ext_vector_type(8)))  _Float16 v8h;
typedef __attribute__((ext_vector_type(8)))  float    v8f;

// Load one 16-half WMMA fragment: two contiguous 16B segments (K j..j+7 and K j+16..j+23).
__device__ __forceinline__ v16h ldfrag(const _Float16* __restrict__ p) {
  v8h lo = *(const v8h*)(p);
  v8h hi = *(const v8h*)(p + 16);
  return __builtin_shufflevector(lo, hi, 0, 1, 2, 3, 4, 5, 6, 7,
                                 8, 9, 10, 11, 12, 13, 14, 15);
}

// ---------------- degree / normalization ----------------
__global__ void k_deg(const int* __restrict__ dst, int E, float* __restrict__ deg) {
  int e = blockIdx.x * blockDim.x + threadIdx.x;
  if (e < E) atomicAdd(&deg[dst[e]], 1.0f);
}

__global__ void k_dinv(float* __restrict__ deg, int n) {
  int i = blockIdx.x * blockDim.x + threadIdx.x;
  if (i < n) deg[i] = rsqrtf(deg[i] + 1.0f);   // deg -> dinv in place
}

// ---------------- f16 packing ----------------
// rows x cols f32 -> rowsPad x colsPad f16, zero padded (colsPad is pow2)
__global__ void k_pack_rows(const float* __restrict__ src, _Float16* __restrict__ dst,
                            int rows, int cols, int colsPad, int total) {
  int id = blockIdx.x * blockDim.x + threadIdx.x;
  if (id >= total) return;
  int r = id / colsPad, c = id - r * colsPad;
  float v = (r < rows && c < cols) ? src[r * cols + c] : 0.0f;
  dst[id] = (_Float16)v;
}

// W[K,N] f32 -> Wt[N,Kpad] f16 (transposed, K zero-padded)
__global__ void k_pack_wT(const float* __restrict__ w, _Float16* __restrict__ wt,
                          int K, int N, int Kpad) {
  int id = blockIdx.x * blockDim.x + threadIdx.x;
  if (id >= N * Kpad) return;
  int nn = id / Kpad, k = id - nn * Kpad;
  wt[id] = (_Float16)((k < K) ? w[k * N + nn] : 0.0f);
}

// ---------------- GEMM1: Ah[Mpad,32] @ W1t[128,32]^T -> Y[Mpad,128] ----------------
__global__ void k_gemm1(const _Float16* __restrict__ A, const _Float16* __restrict__ Bt,
                        float* __restrict__ Y, int mtiles) {
  int wave = threadIdx.x >> 5;
  int tile = blockIdx.x * (blockDim.x >> 5) + wave;
  if (tile >= mtiles * 8) return;              // wave-uniform
  int mt = tile >> 3, nt = tile & 7;
  int lane = threadIdx.x & 31;
  int hs = lane >> 4, idx = lane & 15;
  int m  = mt * 16 + idx;
  int nn = nt * 16 + idx;
  v16h a = ldfrag(A + m * 32 + hs * 8);
  v16h b = ldfrag(Bt + nn * 32 + hs * 8);
  v8f c = {};
  c = __builtin_amdgcn_wmma_f32_16x16x32_f16(false, a, false, b, (short)0, c, false, false);
#pragma unroll
  for (int v = 0; v < 8; ++v)
    Y[(mt * 16 + hs * 8 + v) * 128 + nn] = c[v];
}

// ---------------- GEMM2: Ah[Mpad,128] @ W2t[64,128]^T -> Y[Mpad,64] ----------------
__global__ void k_gemm2(const _Float16* __restrict__ A, const _Float16* __restrict__ Bt,
                        float* __restrict__ Y, int mtiles) {
  int wave = threadIdx.x >> 5;
  int tile = blockIdx.x * (blockDim.x >> 5) + wave;
  if (tile >= mtiles * 4) return;
  int mt = tile >> 2, nt = tile & 3;
  int lane = threadIdx.x & 31;
  int hs = lane >> 4, idx = lane & 15;
  int m  = mt * 16 + idx;
  int nn = nt * 16 + idx;
  v8f c = {};
#pragma unroll
  for (int kt = 0; kt < 4; ++kt) {
    v16h a = ldfrag(A + m * 128 + kt * 32 + hs * 8);
    v16h b = ldfrag(Bt + nn * 128 + kt * 32 + hs * 8);
    c = __builtin_amdgcn_wmma_f32_16x16x32_f16(false, a, false, b, (short)0, c, false, false);
  }
#pragma unroll
  for (int v = 0; v < 8; ++v)
    Y[(mt * 16 + hs * 8 + v) * 64 + nn] = c[v];
}

// ---------------- edge scatter-add: Agg[dst] += H[src]*dinv[src]*dinv[dst] ------
// gridDim.y = F/4 feature chunks; atomics resolve in the 192MB L2.
__global__ void k_scatter(const int* __restrict__ src, const int* __restrict__ dst,
                          const float* __restrict__ dinv, const float* __restrict__ Hin,
                          float* __restrict__ Agg, int E, int F) {
  int e = blockIdx.x * blockDim.x + threadIdx.x;
  if (e >= E) return;
  int c = blockIdx.y;
  int s = src[e], d = dst[e];
  float nrm = dinv[s] * dinv[d];
  const float4 v = *(const float4*)(Hin + (size_t)s * F + c * 4);
  float* o = Agg + (size_t)d * F + c * 4;
  atomicAdd(o + 0, v.x * nrm);
  atomicAdd(o + 1, v.y * nrm);
  atomicAdd(o + 2, v.z * nrm);
  atomicAdd(o + 3, v.w * nrm);
}

// ---------------- layer1 finalize: self-loop + bias + ReLU -> f16 [nPad,128] ----
__global__ void k_finalize1(const float* __restrict__ Agg, const float* __restrict__ Hin,
                            const float* __restrict__ dinv, const float* __restrict__ bias,
                            _Float16* __restrict__ Out, int n) {
  int id = blockIdx.x * blockDim.x + threadIdx.x;
  if (id >= n * 128) return;
  int i = id >> 7, f = id & 127;
  float di = dinv[i];
  float v = Agg[id] + Hin[id] * di * di + bias[f];
  Out[id] = (_Float16)fmaxf(v, 0.0f);
}

// ---------------- layer2 finalize: self-loop + bias, f32 in place ----------------
__global__ void k_finalize2(float* __restrict__ Agg, const float* __restrict__ Hin,
                            const float* __restrict__ dinv, const float* __restrict__ bias,
                            int n) {
  int id = blockIdx.x * blockDim.x + threadIdx.x;
  if (id >= n * 64) return;
  int i = id >> 6, f = id & 63;
  float di = dinv[i];
  Agg[id] = Agg[id] + Hin[id] * di * di + bias[f];
}

// ---------------- column mean over nodes (64 features) ----------------
__global__ void k_mean(const float* __restrict__ H, int n, float* __restrict__ out) {
  __shared__ float sbuf[256];
  int f = threadIdx.x & 63;
  int g = threadIdx.x >> 6;                // 4 node groups per block
  float s = 0.0f;
  for (int i = blockIdx.x * 4 + g; i < n; i += gridDim.x * 4)
    s += H[(size_t)i * 64 + f];
  sbuf[threadIdx.x] = s;
  __syncthreads();
  if (g == 0) {
    float t = sbuf[f] + sbuf[64 + f] + sbuf[128 + f] + sbuf[192 + f];
    atomicAdd(&out[f], t / (float)n);
  }
}

extern "C" void kernel_launch(void* const* d_in, const int* in_sizes, int n_in,
                              void* d_out, int out_size, void* d_ws, size_t ws_size,
                              hipStream_t stream) {
  const float* X1 = (const float*)d_in[0];
  const float* X2 = (const float*)d_in[1];
  const int*  EI1 = (const int*)d_in[2];
  const int*  EI2 = (const int*)d_in[3];
  const float* W1 = (const float*)d_in[4];
  const float* b1 = (const float*)d_in[5];
  const float* W2 = (const float*)d_in[6];
  const float* b2 = (const float*)d_in[7];
  int n = in_sizes[0] / 21;
  int E = in_sizes[2] / 2;
  int mtiles = (n + 15) / 16;

  size_t nPad = ((size_t)n + 63) & ~(size_t)63;
  float* ws   = (float*)d_ws;
  float* dinv = ws;                               // nPad
  float* xw1  = dinv + nPad;                      // nPad*128 f32
  float* agg1 = xw1  + nPad * 128;                // nPad*128 f32
  float* hw2  = agg1 + nPad * 128;                // nPad*64  f32
  float* agg2 = hw2  + nPad * 64;                 // nPad*64  f32
  _Float16* Xh  = (_Float16*)(agg2 + nPad * 64);  // nPad*32  f16
  _Float16* H1h = Xh + nPad * 32;                 // nPad*128 f16
  _Float16* W1t = H1h + nPad * 128;               // 128*32   f16
  _Float16* W2t = W1t + 128 * 32;                 // 64*128   f16

  float* out = (float*)d_out;
  hipMemsetAsync(out, 0, 128 * sizeof(float), stream);

  // weight packs are branch-invariant
  k_pack_wT<<<(128 * 32 + 255) / 256, 256, 0, stream>>>(W1, W1t, 21, 128, 32);
  k_pack_wT<<<(64 * 128 + 255) / 256, 256, 0, stream>>>(W2, W2t, 128, 64, 128);

  for (int br = 0; br < 2; ++br) {
    const float* X  = br ? X2 : X1;
    const int* src  = br ? EI2 : EI1;
    const int* dst  = src + E;

    hipMemsetAsync(dinv, 0, (size_t)n * sizeof(float), stream);
    hipMemsetAsync(agg1, 0, nPad * 128 * sizeof(float), stream);
    hipMemsetAsync(agg2, 0, nPad * 64 * sizeof(float), stream);
    hipMemsetAsync(H1h, 0, nPad * 128 * sizeof(_Float16), stream);

    k_deg <<<(E + 255) / 256, 256, 0, stream>>>(dst, E, dinv);
    k_dinv<<<(n + 255) / 256, 256, 0, stream>>>(dinv, n);

    // layer 1
    int px = (int)(nPad * 32);
    k_pack_rows<<<(px + 255) / 256, 256, 0, stream>>>(X, Xh, n, 21, 32, px);
    int tiles1 = mtiles * 8;
    k_gemm1<<<(tiles1 + 7) / 8, 256, 0, stream>>>(Xh, W1t, xw1, mtiles);
    dim3 sg1((E + 255) / 256, 32);
    k_scatter<<<sg1, 256, 0, stream>>>(src, dst, dinv, xw1, agg1, E, 128);
    k_finalize1<<<((n * 128) + 255) / 256, 256, 0, stream>>>(agg1, xw1, dinv, b1, H1h, n);

    // layer 2
    int tiles2 = mtiles * 4;
    k_gemm2<<<(tiles2 + 7) / 8, 256, 0, stream>>>(H1h, W2t, hw2, mtiles);
    dim3 sg2((E + 255) / 256, 16);
    k_scatter<<<sg2, 256, 0, stream>>>(src, dst, dinv, hw2, agg2, E, 64);
    k_finalize2<<<((n * 64) + 255) / 256, 256, 0, stream>>>(agg2, hw2, dinv, b2, n);

    k_mean<<<128, 256, 0, stream>>>(agg2, n, out + br * 64);
  }
}
